// CrystalHypergraphConv_18064632447537
// MI455X (gfx1250) — matrix operand order repaired and checked by reading, not compiled, not run on hardware
//
#include <hip/hip_runtime.h>
#include <hip/hip_bf16.h>
#include <math.h>

#define N_ATOMS   50000
#define N_MOTIFS  50000
#define N_INC     600000
#define N_GRAPHS  512
#define H_DIM     64
#define MOTIF_DIM 94
#define MOTIF_PAD 96
#define HOUT      128
#define ZDIM      222       // 2*H + MOTIF_DIM
#define ZPAD      224       // padded to 7 chunks of K=32
#define KCHUNKS   7
#define NTILES    4         // 64 output cols / 16
#define WAVES     8         // waves per block in WMMA kernel

typedef _Float16 h8   __attribute__((ext_vector_type(8)));
typedef _Float16 v16h __attribute__((ext_vector_type(16)));
typedef float    v8f  __attribute__((ext_vector_type(8)));

// ---------------------------------------------------------------- utilities
__global__ void k_zero(float* p, int n) {
  int i = blockIdx.x * blockDim.x + threadIdx.x;
  if (i < n) p[i] = 0.0f;
}

// x = embed_table[atom_z], plus f16 copy for WMMA gathers
__global__ void k_embed(const int* __restrict__ atom_z,
                        const float* __restrict__ tbl,
                        float* __restrict__ x32,
                        _Float16* __restrict__ x_h) {
  int i = blockIdx.x * blockDim.x + threadIdx.x;
  if (i >= N_ATOMS * H_DIM) return;
  int a = i >> 6, h = i & 63;
  float v = tbl[atom_z[a] * H_DIM + h];
  x32[i] = v;
  x_h[i] = (_Float16)v;
}

// scatter-sum x[src] into motif accumulators
__global__ void k_hx_acc(const int* __restrict__ src, const int* __restrict__ hid,
                         const float* __restrict__ x32,
                         float* __restrict__ hx_sum, float* __restrict__ hx_cnt) {
  int i = blockIdx.x * blockDim.x + threadIdx.x;
  if (i >= N_INC * H_DIM) return;
  int e = i >> 6, h = i & 63;
  int s = src[e], m = hid[e];
  atomicAdd(&hx_sum[(size_t)m * H_DIM + h], x32[(size_t)s * H_DIM + h]);
  if (h == 0) atomicAdd(&hx_cnt[m], 1.0f);
}

__global__ void k_hx_fin(const float* __restrict__ hx_sum,
                         const float* __restrict__ hx_cnt,
                         _Float16* __restrict__ hx_h) {
  int i = blockIdx.x * blockDim.x + threadIdx.x;
  if (i >= N_MOTIFS * H_DIM) return;
  hx_h[i] = (_Float16)(hx_sum[i] / fmaxf(hx_cnt[i >> 6], 1.0f));
}

// motif_attr -> f16, padded 94 -> 96
__global__ void k_ma_pack(const float* __restrict__ ma, _Float16* __restrict__ ma_h) {
  int i = blockIdx.x * blockDim.x + threadIdx.x;
  if (i >= N_MOTIFS * MOTIF_PAD) return;
  int m = i / MOTIF_PAD, j = i - m * MOTIF_PAD;
  ma_h[i] = (j < MOTIF_DIM) ? (_Float16)ma[(size_t)m * MOTIF_DIM + j] : (_Float16)0.0f;
}

// Pack w_f/w_c into WMMA B-fragment order:
//   wpk[frag][lane][half], frag = (mat*7 + kc)*4 + ntile
//   lane 0-15: N = ntile*16 + lane,   K = kc*32 + half        (half 0..15)
//   lane 16-31: N = ntile*16 + lane-16, K = kc*32 + 16 + half
__global__ void k_wpack(const float* __restrict__ wf, const float* __restrict__ wc,
                        _Float16* __restrict__ wpk) {
  int i = blockIdx.x * blockDim.x + threadIdx.x;
  if (i >= 2 * KCHUNKS * NTILES * 512) return;
  int h    = i & 15;
  int lane = (i >> 4) & 31;
  int frag = i >> 9;
  int mat  = frag / (KCHUNKS * NTILES);
  int rem  = frag % (KCHUNKS * NTILES);
  int kc   = rem >> 2;
  int nt   = rem & 3;
  int K = kc * 32 + h + ((lane >= 16) ? 16 : 0);
  int n = nt * 16 + (lane & 15);
  const float* w = mat ? wc : wf;
  float v = (K < ZDIM) ? w[K * H_DIM + n] : 0.0f;
  wpk[i] = (_Float16)v;
}

// ---------------------------------------------------------------- main WMMA kernel
// One 16-incidence tile per wave. z-tile (16x224 f16) staged in LDS,
// 7 K-chunks x 4 N-tiles x 2 weight matrices = 56 v_wmma_f32_16x16x32_f16.
__global__ void __launch_bounds__(WAVES * 32)
k_msg_wmma(const int* __restrict__ src, const int* __restrict__ hid,
           const _Float16* __restrict__ x_h, const _Float16* __restrict__ hx_h,
           const _Float16* __restrict__ ma_h, const _Float16* __restrict__ wpk,
           const float* __restrict__ b_f, const float* __restrict__ b_c,
           float* __restrict__ out_sum, float* __restrict__ out_cnt) {
  __shared__ __align__(16) _Float16 zs[WAVES][16][ZPAD];

  const int wave = threadIdx.x >> 5;
  const int lane = threadIdx.x & 31;
  const int tile = blockIdx.x * WAVES + wave;
  const int base = tile * 16;

  // ---- stage z rows: 2 lanes per row, vectorized 16B copies
  {
    int r = lane >> 1;
    int e = base + r;
    if (e >= N_INC) e = N_INC - 1;       // clamp; invalid rows never scattered
    int s  = src[e];
    int hd = hid[e];
    h8* pd = (h8*)&zs[wave][r][0];
    if ((lane & 1) == 0) {
      const h8* px = (const h8*)(x_h + (size_t)s * H_DIM);
      #pragma unroll
      for (int q = 0; q < 8; ++q) pd[q] = px[q];          // cols 0..63
      const h8* ph = (const h8*)(hx_h + (size_t)hd * H_DIM);
      #pragma unroll
      for (int q = 0; q < 6; ++q) pd[8 + q] = ph[q];      // cols 64..111
    } else {
      const h8* ph = (const h8*)(hx_h + (size_t)hd * H_DIM);
      #pragma unroll
      for (int q = 6; q < 8; ++q) pd[8 + q] = ph[q];      // cols 112..127
      const h8* pm = (const h8*)(ma_h + (size_t)hd * MOTIF_PAD);
      #pragma unroll
      for (int q = 0; q < 12; ++q) pd[16 + q] = pm[q];    // cols 128..223
    }
  }
  __syncthreads();

  // ---- WMMA accumulation: zf = z @ w_f, zc = z @ w_c  (16 rows x 64 cols each)
  v8f accF[NTILES], accC[NTILES];
  {
    const v8f z0 = {0.f, 0.f, 0.f, 0.f, 0.f, 0.f, 0.f, 0.f};
    #pragma unroll
    for (int t = 0; t < NTILES; ++t) { accF[t] = z0; accC[t] = z0; }
  }
  const int m  = lane & 15;
  const int hi = (lane >= 16) ? 1 : 0;

  #pragma unroll
  for (int kc = 0; kc < KCHUNKS; ++kc) {
    // A fragment (contiguous K order): halves 0..15 = K+0..15 (lo lanes) / K+16..31 (hi lanes),
    // matching wpk B-fragment order
    const _Float16* zrow = &zs[wave][m][kc * 32 + hi * 16];
    union { v16h v; h8 h[2]; } A;
    A.h[0] = *(const h8*)(zrow);
    A.h[1] = *(const h8*)(zrow + 8);

    #pragma unroll
    for (int nt = 0; nt < NTILES; ++nt) {
      const v16h bF = *(const v16h*)(wpk + ((size_t)((0 * KCHUNKS + kc) * NTILES + nt) * 512) + lane * 16);
      const v16h bC = *(const v16h*)(wpk + ((size_t)((1 * KCHUNKS + kc) * NTILES + nt) * 512) + lane * 16);
      accF[nt] = __builtin_amdgcn_wmma_f32_16x16x32_f16(false, A.v, false, bF,
                                                        (short)0, accF[nt], false, false);
      accC[nt] = __builtin_amdgcn_wmma_f32_16x16x32_f16(false, A.v, false, bC,
                                                        (short)0, accC[nt], false, false);
    }
  }

  // ---- gated epilogue + scatter-mean accumulation
  // C/D layout: vgpr v, lane l -> row = v + 8*(l>=16), col = l&15
  const int col_lo = lane & 15;
  const int rowoff = hi * 8;
  #pragma unroll
  for (int nt = 0; nt < NTILES; ++nt) {
    const int col = nt * 16 + col_lo;
    const float bf = b_f[col];
    const float bc = b_c[col];
    #pragma unroll
    for (int v = 0; v < 8; ++v) {
      const int e = base + rowoff + v;
      if (e < N_INC) {
        float zf = accF[nt][v] + bf;
        float zc = accC[nt][v] + bc;
        float sig = 1.0f / (1.0f + __expf(-zf));
        float sp  = (zc > 20.0f) ? zc : log1pf(__expf(zc));
        atomicAdd(&out_sum[(size_t)src[e] * H_DIM + col], sig * sp);
      }
    }
  }
  if (lane < 16) {
    int e = base + lane;
    if (e < N_INC) atomicAdd(&out_cnt[src[e]], 1.0f);
  }
}

// ---------------------------------------------------------------- BN stats: one block per column
__global__ void k_colstats(float* __restrict__ out_sum, const float* __restrict__ out_cnt,
                           const float* __restrict__ gamma, const float* __restrict__ beta,
                           float* __restrict__ bn_scale, float* __restrict__ bn_shift) {
  __shared__ float sh1[256];
  __shared__ float sh2[256];
  const int col = blockIdx.x;       // 0..63
  const int t = threadIdx.x;
  float s = 0.f, s2 = 0.f;
  for (int a = t; a < N_ATOMS; a += 256) {
    float o = out_sum[(size_t)a * H_DIM + col] / fmaxf(out_cnt[a], 1.0f);
    out_sum[(size_t)a * H_DIM + col] = o;     // normalize in place
    s += o; s2 += o * o;
  }
  sh1[t] = s; sh2[t] = s2;
  __syncthreads();
  for (int st = 128; st > 0; st >>= 1) {
    if (t < st) { sh1[t] += sh1[t + st]; sh2[t] += sh2[t + st]; }
    __syncthreads();
  }
  if (t == 0) {
    float mu  = sh1[0] / (float)N_ATOMS;
    float var = fmaxf(sh2[0] / (float)N_ATOMS - mu * mu, 0.0f);
    float sc  = gamma[col] * rsqrtf(var + 1e-5f);
    bn_scale[col] = sc;
    bn_shift[col] = beta[col] - mu * sc;
  }
}

// BN + residual + relu + scatter-mean pooling over graphs
__global__ void k_bnpool(const float* __restrict__ out_sum, const float* __restrict__ x32,
                         const int* __restrict__ batch,
                         const float* __restrict__ bn_scale, const float* __restrict__ bn_shift,
                         float* __restrict__ g_sum, float* __restrict__ g_cnt) {
  int i = blockIdx.x * blockDim.x + threadIdx.x;
  if (i >= N_ATOMS * H_DIM) return;
  int a = i >> 6, h = i & 63;
  float v = fmaxf(out_sum[i] * bn_scale[h] + bn_shift[h] + x32[i], 0.0f);
  int b = batch[a];
  atomicAdd(&g_sum[(size_t)b * H_DIM + h], v);
  if (h == 0) atomicAdd(&g_cnt[b], 1.0f);
}

// MLP head: one block (128 threads) per graph
__global__ void k_head(const float* __restrict__ g_sum, const float* __restrict__ g_cnt,
                       const float* __restrict__ w_l1, const float* __restrict__ b_l1,
                       const float* __restrict__ w_out, const float* __restrict__ b_out,
                       float* __restrict__ out) {
  __shared__ float gsh[H_DIM];
  __shared__ float red[HOUT];
  const int g = blockIdx.x;
  const int t = threadIdx.x;
  if (t < H_DIM) gsh[t] = g_sum[(size_t)g * H_DIM + t] / fmaxf(g_cnt[g], 1.0f);
  __syncthreads();
  float acc = b_l1[t];
  #pragma unroll 8
  for (int k = 0; k < H_DIM; ++k) acc += gsh[k] * w_l1[k * HOUT + t];
  float hv = (acc > 20.0f) ? acc : log1pf(__expf(acc));
  red[t] = hv * w_out[t];
  __syncthreads();
  for (int st = 64; st > 0; st >>= 1) {
    if (t < st) red[t] += red[t + st];
    __syncthreads();
  }
  if (t == 0) out[g] = red[0] + b_out[0];
}

// ---------------------------------------------------------------- launcher
extern "C" void kernel_launch(void* const* d_in, const int* in_sizes, int n_in,
                              void* d_out, int out_size, void* d_ws, size_t ws_size,
                              hipStream_t stream) {
  const int*   atom_z     = (const int*)d_in[0];
  const float* motif_attr = (const float*)d_in[1];
  const int*   hei        = (const int*)d_in[2];   // [2, N_INC] row-major
  const int*   batch      = (const int*)d_in[3];
  const float* embed_tbl  = (const float*)d_in[4];
  const float* w_f        = (const float*)d_in[5];
  const float* b_f        = (const float*)d_in[6];
  const float* w_c        = (const float*)d_in[7];
  const float* b_c        = (const float*)d_in[8];
  const float* bn_gamma   = (const float*)d_in[9];
  const float* bn_beta    = (const float*)d_in[10];
  const float* w_l1       = (const float*)d_in[11];
  const float* b_l1       = (const float*)d_in[12];
  const float* w_out      = (const float*)d_in[13];
  const float* b_out      = (const float*)d_in[14];
  float* out = (float*)d_out;

  const int* src = hei;
  const int* hid = hei + N_INC;

  // ---- workspace carve (floats first; zero-region is contiguous at the front)
  float* ws_f = (float*)d_ws;
  size_t o = 0;
  float* hx_sum  = ws_f + o; o += (size_t)N_MOTIFS * H_DIM;
  float* hx_cnt  = ws_f + o; o += N_MOTIFS;
  float* out_sum = ws_f + o; o += (size_t)N_ATOMS * H_DIM;
  float* out_cnt = ws_f + o; o += N_ATOMS;
  float* g_sum   = ws_f + o; o += (size_t)N_GRAPHS * H_DIM;
  float* g_cnt   = ws_f + o; o += N_GRAPHS;
  const int ZERO_N = (int)o;                         // 6,533,280
  float* bn_scale = ws_f + o; o += H_DIM;
  float* bn_shift = ws_f + o; o += H_DIM;
  float* x32      = ws_f + o; o += (size_t)N_ATOMS * H_DIM;

  _Float16* hbase = (_Float16*)(ws_f + o);
  size_t ho = 0;
  _Float16* x_h  = hbase + ho; ho += (size_t)N_ATOMS * H_DIM;
  _Float16* hx_h = hbase + ho; ho += (size_t)N_MOTIFS * H_DIM;
  _Float16* ma_h = hbase + ho; ho += (size_t)N_MOTIFS * MOTIF_PAD;
  ho = (ho + 15) & ~(size_t)15;                      // 32B align for v16h loads
  _Float16* wpk  = hbase + ho; ho += (size_t)2 * KCHUNKS * NTILES * 512;

  const int B = 256;
  // 1) zero accumulators
  k_zero<<<(ZERO_N + B - 1) / B, B, 0, stream>>>(ws_f, ZERO_N);
  // 2) embedding lookup (f32 + f16)
  k_embed<<<(N_ATOMS * H_DIM + B - 1) / B, B, 0, stream>>>(atom_z, embed_tbl, x32, x_h);
  // 3) scatter x into motif accumulators
  k_hx_acc<<<(N_INC * H_DIM + B - 1) / B, B, 0, stream>>>(src, hid, x32, hx_sum, hx_cnt);
  // 4) hx mean -> f16; motif_attr -> padded f16
  k_hx_fin<<<(N_MOTIFS * H_DIM + B - 1) / B, B, 0, stream>>>(hx_sum, hx_cnt, hx_h);
  k_ma_pack<<<(N_MOTIFS * MOTIF_PAD + B - 1) / B, B, 0, stream>>>(motif_attr, ma_h);
  // 5) weights -> B-fragment layout
  k_wpack<<<(2 * KCHUNKS * NTILES * 512 + B - 1) / B, B, 0, stream>>>(w_f, w_c, wpk);
  // 6) main WMMA message kernel
  {
    const int ntiles = (N_INC + 15) / 16;            // 37500
    const int grid = (ntiles + WAVES - 1) / WAVES;   // 4688
    k_msg_wmma<<<grid, WAVES * 32, 0, stream>>>(src, hid, x_h, hx_h, ma_h, wpk,
                                                b_f, b_c, out_sum, out_cnt);
  }
  // 7) per-column batch-norm stats (normalizes out_sum in place)
  k_colstats<<<H_DIM, 256, 0, stream>>>(out_sum, out_cnt, bn_gamma, bn_beta,
                                        bn_scale, bn_shift);
  // 8) BN + residual + relu + graph pooling
  k_bnpool<<<(N_ATOMS * H_DIM + B - 1) / B, B, 0, stream>>>(out_sum, x32, batch,
                                                            bn_scale, bn_shift, g_sum, g_cnt);
  // 9) MLP head
  k_head<<<N_GRAPHS, HOUT, 0, stream>>>(g_sum, g_cnt, w_l1, b_l1, w_out, b_out, out);

  (void)in_sizes; (void)n_in; (void)out_size; (void)ws_size;
}
